// GraphTransformer_30374008717979
// MI455X (gfx1250) — matrix operand order repaired and checked by reading, not compiled, not run on hardware
//
#include <hip/hip_runtime.h>

// ---------------------------------------------------------------------------
// GraphTransformer (TransformerConv + Linear + ReLU + BatchNorm) for gfx1250.
// N=50000 nodes, E=640000 edges, IN=128, OUT=32, H=5 heads.
//
// Pipeline (all on `stream`):
//   0. init         : zero atomics buffers, seed segment-max with -inf codes
//   1. pack_w       : pack [Wq|Wk|Wv|Wskip] (128x512) into WMMA B-fragment
//                     layout (bf16) + concatenated bias
//   2. qkvs_gemm    : qkvs[N,512] = x @ Wcat + bcat via v_wmma_f32_16x16x32_bf16;
//                     x tile staged once per block into LDS as bf16 (padded,
//                     bank-conflict-free), A-fragments via ds_load_b128
//   3. edge_scores  : wave/edge, 5 head dot-products + ordered-int atomicMax
//   4. edge_exp     : ex = exp(score - max), atomicAdd denominators
//   5. edge_agg     : wave/edge scatter-add of (1/H)*sum_h alpha_h * v  (f32 atomics)
//   6. head_pass    : h = relu((agg+skip) @ W1 + b1) via WMMA (K=32), fused
//                     per-channel sum/sumsq reduction for BatchNorm
//   7. bn_apply     : y = (h - mu) * rsqrt(var+eps) * gamma + beta (in place)
// ---------------------------------------------------------------------------

#define N_NODES   50000
#define N_EDGES   640000
#define IN_CH     128
#define OUT_CH    32
#define HEADS     5
#define QCOLS     512           // 160 q | 160 k | 160 v | 32 skip
#define BN_EPS    1e-5f
#define ROW_TILES (N_NODES / 16)   // 3125 (exact)
#define XT_PITCH  136              // 128 + 8 bf16 pad -> 272B row stride (16B mult,
                                   // shifts LDS bank by 4 per row: conflict-free)

typedef __bf16 v16bf __attribute__((ext_vector_type(16)));
typedef __bf16 v8bf  __attribute__((ext_vector_type(8)));
typedef __bf16 v4bf  __attribute__((ext_vector_type(4)));
typedef float  v8f   __attribute__((ext_vector_type(8)));
typedef float  v4f   __attribute__((ext_vector_type(4)));

// f32 -> bf16 (native v_cvt path, RNE)
__device__ __forceinline__ __bf16 f2bf(float f) { return (__bf16)f; }

// 16-bit A-fragment (16x32) K index for slot i, lane-group g  (ISA 7.12.2)
__device__ __forceinline__ int kpatA(int i, int g) {
    return (i < 8) ? (g * 8 + i) : (16 + g * 8 + (i - 8));
}
// 16-bit B-fragment (32x16) K index for slot i, lane-group g
__device__ __forceinline__ int kpatB(int i, int g) {
    return (i < 8) ? (g * 8 + i) : (16 + g * 8 + (i - 8));
}

// order-preserving float<->int for atomicMax on signed ints
__device__ __forceinline__ int orderFloat(float f) {
    int i = __float_as_int(f);
    return (i >= 0) ? i : (i ^ 0x7fffffff);
}
__device__ __forceinline__ float unorderFloat(int i) {
    return __int_as_float((i >= 0) ? i : (i ^ 0x7fffffff));
}

// ---------------------------------------------------------------------------
__global__ void init_buffers(float* aggm, float* denom, int* mbuf, float* chans) {
    int idx = blockIdx.x * blockDim.x + threadIdx.x;
    if (idx < N_NODES * OUT_CH) aggm[idx] = 0.0f;
    if (idx < N_NODES * HEADS) { denom[idx] = 0.0f; mbuf[idx] = (int)0x80000000; }
    if (idx < 64) chans[idx] = 0.0f;
}

// ---------------------------------------------------------------------------
// Pack W columns [0,160)=Wq [160,320)=Wk [320,480)=Wv [480,512)=Wskip into the
// exact bf16 B-fragment layout: flat index = ((ct*4+kk)*32 + lane)*16 + slot.
__global__ void pack_w(const float* __restrict__ Wq, const float* __restrict__ Wk,
                       const float* __restrict__ Wv, const float* __restrict__ Ws,
                       const float* __restrict__ bq, const float* __restrict__ bk,
                       const float* __restrict__ bv, const float* __restrict__ bs,
                       __bf16* __restrict__ wpack, float* __restrict__ bcat) {
    int idx = blockIdx.x * blockDim.x + threadIdx.x;
    if (idx < QCOLS) {
        int c = idx; float b;
        if (c < 160)      b = bq[c];
        else if (c < 320) b = bk[c - 160];
        else if (c < 480) b = bv[c - 320];
        else              b = bs[c - 480];
        bcat[c] = b;
    }
    if (idx >= 32 * 4 * 32 * 16) return;
    int i    = idx & 15;
    int lane = (idx >> 4) & 31;
    int kk   = (idx >> 9) & 3;
    int ct   = idx >> 11;                 // 0..31 column tile
    int g = lane >> 4, n = lane & 15;
    int k   = kk * 32 + kpatB(i, g);      // 0..127
    int col = ct * 16 + n;                // 0..511
    float w;
    if (col < 160)      w = Wq[k * 160 + col];
    else if (col < 320) w = Wk[k * 160 + (col - 160)];
    else if (col < 480) w = Wv[k * 160 + (col - 320)];
    else                w = Ws[k * 32  + (col - 480)];
    wpack[idx] = f2bf(w);
}

// ---------------------------------------------------------------------------
// qkvs[N,512] = x[N,128] @ Wcat[128,512] + bcat.  One block (8 waves) per
// 16-row strip; each wave owns 64 output columns (4 WMMA accumulators).
// x tile is staged once per block into LDS as bf16 (converted at stage time).
__global__ __launch_bounds__(256) void qkvs_gemm(const float* __restrict__ x,
                                                 const __bf16* __restrict__ wpack,
                                                 const float* __restrict__ bcat,
                                                 float* __restrict__ qkvs) {
    __shared__ __bf16 xt[16 * XT_PITCH];

    const int lane = threadIdx.x & 31;
    const int wave = threadIdx.x >> 5;        // 0..7 -> column group
    const int rt   = blockIdx.x;              // 0..3124 row tile
    const int g = lane >> 4, n = lane & 15;

    // cooperative stage: 256 threads x 8 floats = full 16x128 tile, one pass
    {
        int row = threadIdx.x >> 4;           // 0..15
        int col = (threadIdx.x & 15) * 8;     // 0,8,...,120
        const float* src = x + (size_t)(rt * 16 + row) * IN_CH + col;
        v4f f0 = *(const v4f*)(src);
        v4f f1 = *(const v4f*)(src + 4);
        v4bf c0 = __builtin_convertvector(f0, v4bf);   // packed cvt f32->bf16
        v4bf c1 = __builtin_convertvector(f1, v4bf);
        v8bf c  = __builtin_shufflevector(c0, c1, 0, 1, 2, 3, 4, 5, 6, 7);
        *(v8bf*)(xt + row * XT_PITCH + col) = c;       // 16B ds_store
    }
    __syncthreads();

    v8f acc[4] = {};

    #pragma unroll
    for (int kk = 0; kk < 4; ++kk) {          // K chunks of 32
        // A-fragment: row = n, slots 0..7 = K kk*32+g*8.., slots 8..15 = +16
        const __bf16* ap = xt + n * XT_PITCH + kk * 32 + g * 8;
        v8bf lo = *(const v8bf*)(ap);                  // ds_load_b128
        v8bf hi = *(const v8bf*)(ap + 16);             // ds_load_b128
        v16bf a = __builtin_shufflevector(lo, hi, 0, 1, 2, 3, 4, 5, 6, 7,
                                                  8, 9, 10, 11, 12, 13, 14, 15);
        #pragma unroll
        for (int t = 0; t < 4; ++t) {
            int ct = wave * 4 + t;
            v16bf b = *(const v16bf*)(wpack + ((size_t)(ct * 4 + kk) * 32 + lane) * 16);
            acc[t] = __builtin_amdgcn_wmma_f32_16x16x32_bf16(
                false, a, false, b, (short)0, acc[t], false, false);
        }
    }
    #pragma unroll
    for (int t = 0; t < 4; ++t) {
        int col = wave * 64 + t * 16 + n;
        float bias = bcat[col];
        #pragma unroll
        for (int j = 0; j < 8; ++j) {
            int row = rt * 16 + g * 8 + j;    // C/D layout: M = g*8 + j, N = n
            qkvs[(size_t)row * QCOLS + col] = acc[t][j] + bias;
        }
    }
}

// ---------------------------------------------------------------------------
// One wave per edge; lane = feature dim d (0..31), loop over 5 heads,
// cross-lane reduce, ordered-int atomicMax per (dst, head).
__global__ __launch_bounds__(256) void edge_scores(const int* __restrict__ ei,
                                                   const float* __restrict__ qkvs,
                                                   float* __restrict__ sbuf,
                                                   int* __restrict__ mbuf) {
    int w    = (blockIdx.x * blockDim.x + threadIdx.x) >> 5;
    int lane = threadIdx.x & 31;
    if (w >= N_EDGES) return;
    int src = ei[w];
    int dst = ei[N_EDGES + w];
    const float* q = qkvs + (size_t)dst * QCOLS;          // query at dst
    const float* k = qkvs + (size_t)src * QCOLS + 160;    // key at src
    const float scale = 0.17677669529663687f;             // 1/sqrt(32)
    #pragma unroll
    for (int h = 0; h < HEADS; ++h) {
        float p = q[h * 32 + lane] * k[h * 32 + lane];
        #pragma unroll
        for (int off = 16; off; off >>= 1) p += __shfl_xor(p, off);
        p *= scale;
        if (lane == 0) {
            sbuf[(size_t)w * HEADS + h] = p;
            atomicMax(&mbuf[dst * HEADS + h], orderFloat(p));
        }
    }
}

// ---------------------------------------------------------------------------
__global__ void edge_exp(const int* __restrict__ ei, float* __restrict__ sbuf,
                         const int* __restrict__ mbuf, float* __restrict__ denom) {
    int idx = blockIdx.x * blockDim.x + threadIdx.x;
    if (idx >= N_EDGES * HEADS) return;
    int e = idx / HEADS, h = idx - e * HEADS;
    int dst = ei[N_EDGES + e];
    float m  = unorderFloat(mbuf[dst * HEADS + h]);
    float ex = __expf(sbuf[idx] - m);
    sbuf[idx] = ex;                                       // scores -> exp in place
    atomicAdd(&denom[dst * HEADS + h], ex);
}

// ---------------------------------------------------------------------------
// One wave per edge: aggm[dst, d] += (1/H) * sum_h alpha_h * v[src, h, d]
__global__ __launch_bounds__(256) void edge_agg(const int* __restrict__ ei,
                                                const float* __restrict__ qkvs,
                                                const float* __restrict__ sbuf,
                                                const float* __restrict__ denom,
                                                float* __restrict__ aggm) {
    int w    = (blockIdx.x * blockDim.x + threadIdx.x) >> 5;
    int lane = threadIdx.x & 31;
    if (w >= N_EDGES) return;
    int src = ei[w];
    int dst = ei[N_EDGES + w];
    const float* v = qkvs + (size_t)src * QCOLS + 320;    // value at src
    float val = 0.0f;
    #pragma unroll
    for (int h = 0; h < HEADS; ++h) {
        float dn = denom[dst * HEADS + h];
        dn = dn > 0.0f ? dn : 1.0f;
        float alpha = sbuf[(size_t)w * HEADS + h] / dn;
        val += alpha * v[h * 32 + lane];
    }
    atomicAdd(&aggm[(size_t)dst * OUT_CH + lane], 0.2f * val);
}

// ---------------------------------------------------------------------------
// h = relu((agg + skip) @ W1 + b1) via WMMA (single K=32 step), write h to out,
// reduce per-channel sum/sumsq into chans[0..31]/chans[32..63].
__global__ __launch_bounds__(256) void head_pass(const float* __restrict__ aggm,
                                                 const float* __restrict__ qkvs,
                                                 const float* __restrict__ W1,
                                                 const float* __restrict__ b1,
                                                 float* __restrict__ out,
                                                 float* __restrict__ chans) {
    __shared__ float lsum[32], lssq[32];
    int lane = threadIdx.x & 31, wave = threadIdx.x >> 5;
    if (threadIdx.x < 32) { lsum[threadIdx.x] = 0.0f; lssq[threadIdx.x] = 0.0f; }
    __syncthreads();

    int tile = blockIdx.x * 8 + wave;
    int g = lane >> 4, n = lane & 15;
    if (tile < ROW_TILES) {                    // wave-uniform (EXEC all-ones inside)
        int r0 = tile * 16 + n;
        v16bf a;
        #pragma unroll
        for (int i = 0; i < 16; ++i) {
            int k = kpatA(i, g);               // K = input channel 0..31
            float vv = aggm[(size_t)r0 * OUT_CH + k]
                     + qkvs[(size_t)r0 * QCOLS + 480 + k];   // skip connection
            a[i] = f2bf(vv);
        }
        #pragma unroll
        for (int t = 0; t < 2; ++t) {
            v16bf b;
            #pragma unroll
            for (int i = 0; i < 16; ++i) {
                int k = kpatB(i, g);
                b[i] = f2bf(W1[k * OUT_CH + t * 16 + n]);
            }
            v8f c = {};
            c = __builtin_amdgcn_wmma_f32_16x16x32_bf16(
                false, a, false, b, (short)0, c, false, false);
            int col = t * 16 + n;
            float bias = b1[col];
            float s = 0.0f, ss = 0.0f;
            #pragma unroll
            for (int j = 0; j < 8; ++j) {
                int row = tile * 16 + g * 8 + j;
                float hv = c[j] + bias;
                hv = hv > 0.0f ? hv : 0.0f;
                out[(size_t)row * OUT_CH + col] = hv;
                s += hv; ss += hv * hv;
            }
            atomicAdd(&lsum[col], s);
            atomicAdd(&lssq[col], ss);
        }
    }
    __syncthreads();
    if (threadIdx.x < 32) {
        atomicAdd(&chans[threadIdx.x],      lsum[threadIdx.x]);
        atomicAdd(&chans[32 + threadIdx.x], lssq[threadIdx.x]);
    }
}

// ---------------------------------------------------------------------------
__global__ void bn_apply(float* __restrict__ out, const float* __restrict__ chans,
                         const float* __restrict__ gamma, const float* __restrict__ beta) {
    int idx = blockIdx.x * blockDim.x + threadIdx.x;
    if (idx >= N_NODES * OUT_CH) return;
    int ch = idx & 31;
    const float invN = 1.0f / (float)N_NODES;
    float mu  = chans[ch] * invN;
    float var = chans[32 + ch] * invN - mu * mu;
    float inv = rsqrtf(var + BN_EPS) * gamma[ch];
    out[idx] = (out[idx] - mu) * inv + beta[ch];
}

// ---------------------------------------------------------------------------
extern "C" void kernel_launch(void* const* d_in, const int* in_sizes, int n_in,
                              void* d_out, int out_size, void* d_ws, size_t ws_size,
                              hipStream_t stream) {
    const float* x     = (const float*)d_in[0];
    const int*   ei    = (const int*)d_in[1];     // [2, E]; row0 = src, row1 = dst
    // d_in[2] = edge_attr: unused (matches reference)
    const float* Wq    = (const float*)d_in[3];
    const float* bq    = (const float*)d_in[4];
    const float* Wk    = (const float*)d_in[5];
    const float* bk    = (const float*)d_in[6];
    const float* Wv    = (const float*)d_in[7];
    const float* bv    = (const float*)d_in[8];
    const float* Wskip = (const float*)d_in[9];
    const float* bskip = (const float*)d_in[10];
    const float* W1    = (const float*)d_in[11];
    const float* b1    = (const float*)d_in[12];
    const float* gamma = (const float*)d_in[13];
    const float* beta  = (const float*)d_in[14];
    float* out = (float*)d_out;

    // workspace layout (float units)
    float*  ws    = (float*)d_ws;
    __bf16* wpack = (__bf16*)ws;                                   // 65536 bf16 = 32768 f
    float*  bcat  = ws + 32768;                                    // 512
    float*  qkvs  = bcat + 512;                                    // N*512
    float*  sbuf  = qkvs + (size_t)N_NODES * QCOLS;                // E*5  (scores -> exp)
    int*    mbuf  = (int*)(sbuf + (size_t)N_EDGES * HEADS);        // N*5  ordered-int max
    float*  denom = (float*)(mbuf + N_NODES * HEADS);              // N*5
    float*  aggm  = denom + N_NODES * HEADS;                       // N*32
    float*  chans = aggm + (size_t)N_NODES * OUT_CH;               // 64 (sum | sumsq)

    const int TB = 256;

    init_buffers<<<(N_NODES * OUT_CH + TB - 1) / TB, TB, 0, stream>>>(aggm, denom, mbuf, chans);
    pack_w<<<(32 * 4 * 32 * 16) / TB, TB, 0, stream>>>(Wq, Wk, Wv, Wskip, bq, bk, bv, bskip,
                                                       wpack, bcat);
    qkvs_gemm<<<ROW_TILES, TB, 0, stream>>>(x, wpack, bcat, qkvs);
    edge_scores<<<(N_EDGES * 32 + TB - 1) / TB, TB, 0, stream>>>(ei, qkvs, sbuf, mbuf);
    edge_exp<<<(N_EDGES * HEADS + TB - 1) / TB, TB, 0, stream>>>(ei, sbuf, mbuf, denom);
    edge_agg<<<(N_EDGES * 32 + TB - 1) / TB, TB, 0, stream>>>(ei, qkvs, sbuf, denom, aggm);
    head_pass<<<(ROW_TILES + 7) / 8, TB, 0, stream>>>(aggm, qkvs, W1, b1, out, chans);
    bn_apply<<<(N_NODES * OUT_CH + TB - 1) / TB, TB, 0, stream>>>(out, chans, gamma, beta);

    (void)in_sizes; (void)n_in; (void)out_size; (void)ws_size;
}
